// PINN_66640712565422
// MI455X (gfx1250) — matrix operand order repaired and checked by reading, not compiled, not run on hardware
//
#include <hip/hip_runtime.h>
#include <math.h>

// ---------------------------------------------------------------------------
// PINN 2nd-order jet forward (x, dx/dt, d2x/dt2) for MI455X / gfx1250.
//
// Compute-bound (AI ~12,000 FLOP/B): all hidden-layer matmuls run as
// v_wmma_f32_16x16x32_f16 (f16 operands, f32 accumulate); tanh / LayerNorm /
// jet combination stay fp32. One wave = one 16-sample tile; 3 jet channels
// (value, d/dt, d2/dt2) share each B (weight) fragment. Blocks are persistent
// over TILES sample-tiles so the 66 KB LDS weight staging is amortized 4x.
// ---------------------------------------------------------------------------

typedef __attribute__((ext_vector_type(16))) _Float16 v16h;
typedef __attribute__((ext_vector_type(8)))  float    v8f;

#define LN_EPS 1e-5f
#define TW 128            // padded feature stride of per-wave tile buffers
#define TILES 4           // sample tiles processed per block (per wave)

union HFrag { v16h v; unsigned u[8]; };

__device__ __forceinline__ unsigned ld_u32(const void* p) {
  unsigned r; __builtin_memcpy(&r, p, 4); return r;   // may-alias safe LDS load
}

// ---- 3-channel GEMM: U[c] (16 x FO, f32) = A[c] (16 x FI, f16) * W^T -------
// W lives in LDS f16 row-major [FO][FI]; B fragment element (K,N) = W[N][K].
// Fragment layouts per CDNA5 ISA §7.12.2 (wave32):
//   A (16x32 f16): lane = M&15; lanes 0-15 hold K 0..7,16..23; lanes 16-31
//                  hold K 8..15,24..31; VGPR v packs K = 2*(v&3)+(v>=4?16:0)(+base).
//   B (32x16 f16): lane = N&15; same K order as A: lanes 0-15 hold K 0..7,16..23;
//                  lanes 16-31 hold K 8..15,24..31; VGPR v packs K = 2*(v&3)+(v>=4?16:0)(+base).
//   C (16x16 f32): VGPR j = row M = j + 8*(lane>=16); lane&15 = N.
template<int FI, int FO>
__device__ __forceinline__ void gemm3(const _Float16* __restrict__ Wl,
                                      const _Float16* A0, const _Float16* A1,
                                      const _Float16* A2,
                                      float* U0, float* U1, float* U2, int lane)
{
  const int ln15 = lane & 15;
  const int hi   = lane >> 4;
  const int mrow = hi * 8;
  #pragma unroll
  for (int nt = 0; nt < FO / 16; ++nt) {
    v8f acc0 = {}; v8f acc1 = {}; v8f acc2 = {};
    #pragma unroll
    for (int kt = 0; kt < FI / 32; ++kt) {
      // ---- B fragment (shared by all 3 jet channels) ----
      HFrag bf;
      const _Float16* wrow = Wl + (nt * 16 + ln15) * FI + kt * 32 + hi * 8;
      #pragma unroll
      for (int v = 0; v < 8; ++v) bf.u[v] = ld_u32(wrow + 2 * (v & 3) + ((v >= 4) ? 16 : 0));
      // ---- A fragments, one per channel ----
      const int ab = kt * 32 + hi * 8;
      HFrag a0, a1, a2;
      #pragma unroll
      for (int v = 0; v < 8; ++v) {
        const int K = ab + 2 * (v & 3) + ((v >= 4) ? 16 : 0);
        a0.u[v] = ld_u32(A0 + ln15 * TW + K);
        a1.u[v] = ld_u32(A1 + ln15 * TW + K);
        a2.u[v] = ld_u32(A2 + ln15 * TW + K);
      }
      acc0 = __builtin_amdgcn_wmma_f32_16x16x32_f16(false, a0.v, false, bf.v,
                                                    (short)0, acc0, false, false);
      acc1 = __builtin_amdgcn_wmma_f32_16x16x32_f16(false, a1.v, false, bf.v,
                                                    (short)0, acc1, false, false);
      acc2 = __builtin_amdgcn_wmma_f32_16x16x32_f16(false, a2.v, false, bf.v,
                                                    (short)0, acc2, false, false);
    }
    #pragma unroll
    for (int j = 0; j < 8; ++j) {
      const int off = (mrow + j) * TW + nt * 16 + ln15;
      U0[off] = acc0[j]; U1[off] = acc1[j]; U2[off] = acc2[j];
    }
  }
}

// ---- jet tanh + jet LayerNorm over width W (fp32), writes f16 A for next GEMM
// lane owns row m = lane&15, columns [half*W/2, half*W/2 + W/2); row stats are
// finished with one __shfl_xor across the lane pair (same wave, no barrier).
template<int W>
__device__ __forceinline__ void tanh_ln(float* U0, float* U1, float* U2,
                                        _Float16* A0, _Float16* A1, _Float16* A2,
                                        const float* bias, const float* gam,
                                        const float* bet, int lane)
{
  const int m  = lane & 15;
  const int hh = lane >> 4;
  const int j0 = hh * (W / 2);
  const float inv = 1.0f / (float)W;

  // pass 1: u -> y = tanh jet, accumulate means
  float s0 = 0.f, s1 = 0.f, s2 = 0.f;
  #pragma unroll 4
  for (int j = j0; j < j0 + W / 2; ++j) {
    const int o = m * TW + j;
    const float u0 = U0[o] + bias[j];
    const float u1 = U1[o];
    const float u2 = U2[o];
    const float y0 = tanhf(u0);
    const float sh = 1.0f - y0 * y0;            // sech^2
    const float y1 = sh * u1;
    const float y2 = sh * u2 - 2.0f * y0 * y1 * u1;
    U0[o] = y0; U1[o] = y1; U2[o] = y2;
    s0 += y0; s1 += y1; s2 += y2;
  }
  s0 += __shfl_xor(s0, 16); s1 += __shfl_xor(s1, 16); s2 += __shfl_xor(s2, 16);
  const float mu0 = s0 * inv, mu1 = s1 * inv, mu2 = s2 * inv;

  // pass 2: centered second moments of the jet
  float v00 = 0.f, v01 = 0.f, v11 = 0.f, v02 = 0.f;
  #pragma unroll 4
  for (int j = j0; j < j0 + W / 2; ++j) {
    const int o = m * TW + j;
    const float c0 = U0[o] - mu0;
    const float c1 = U1[o] - mu1;
    const float c2 = U2[o] - mu2;
    v00 += c0 * c0; v01 += c0 * c1; v11 += c1 * c1; v02 += c0 * c2;
  }
  v00 += __shfl_xor(v00, 16); v01 += __shfl_xor(v01, 16);
  v11 += __shfl_xor(v11, 16); v02 += __shfl_xor(v02, 16);
  const float var0 = v00 * inv;
  const float var1 = 2.0f * v01 * inv;
  const float var2 = 2.0f * (v11 + v02) * inv;
  const float r  = rsqrtf(var0 + LN_EPS);
  const float r3 = r * r * r;
  const float r1 = -0.5f * r3 * var1;
  const float r2 = -0.5f * (3.0f * r * r * r1 * var1 + r3 * var2);

  // pass 3: z jet -> fp32 (for final dot) and f16 (next-layer WMMA A operand)
  #pragma unroll 4
  for (int j = j0; j < j0 + W / 2; ++j) {
    const int o = m * TW + j;
    const float c0 = U0[o] - mu0;
    const float c1 = U1[o] - mu1;
    const float c2 = U2[o] - mu2;
    const float g = gam[j], b = bet[j];
    const float z0 = c0 * r * g + b;
    const float z1 = (c1 * r + c0 * r1) * g;
    const float z2 = (c2 * r + 2.0f * c1 * r1 + c0 * r2) * g;
    U0[o] = z0; U1[o] = z1; U2[o] = z2;
    A0[o] = (_Float16)z0; A1[o] = (_Float16)z1; A2[o] = (_Float16)z2;
  }
}

// ---------------- LDS layout (dynamic, ~213 KB < 320 KB/WGP) ----------------
#define OFF_W1   0                      // 128*64  f16 = 16384 B
#define OFF_W2   16384                  // 128*128 f16 = 32768 B
#define OFF_W3   49152                  // 64*128  f16 = 16384 B
#define OFF_W0   65536                  // 64 f32
#define OFF_W4   65792                  // 64 f32
#define OFF_BIAS 66048                  // 384 f32 (b0..b3 packed 0/64/192/320)
#define OFF_GAM  67584                  // 384 f32
#define OFF_BET  69120                  // 384 f32
#define OFF_B4   70656                  // 1 f32 (+pad)
#define OFF_U    70912                  // 4 waves * 3 ch * 16*128 f32 = 98304 B
#define OFF_A    169216                 // 4 waves * 3 ch * 16*128 f16 = 49152 B
#define SMEM_BYTES 218368

__global__ __launch_bounds__(128, 1)
void pinn_jet2_kernel(const float* __restrict__ t,
                      const float* __restrict__ W0, const float* __restrict__ b0,
                      const float* __restrict__ g0, const float* __restrict__ be0,
                      const float* __restrict__ W1, const float* __restrict__ b1,
                      const float* __restrict__ g1, const float* __restrict__ be1,
                      const float* __restrict__ W2, const float* __restrict__ b2,
                      const float* __restrict__ g2, const float* __restrict__ be2,
                      const float* __restrict__ W3, const float* __restrict__ b3,
                      const float* __restrict__ g3, const float* __restrict__ be3,
                      const float* __restrict__ W4, const float* __restrict__ b4,
                      float* __restrict__ out, int B)
{
  extern __shared__ char smem[];
  _Float16* w1l   = (_Float16*)(smem + OFF_W1);
  _Float16* w2l   = (_Float16*)(smem + OFF_W2);
  _Float16* w3l   = (_Float16*)(smem + OFF_W3);
  float*    w0l   = (float*)(smem + OFF_W0);
  float*    w4l   = (float*)(smem + OFF_W4);
  float*    biasl = (float*)(smem + OFF_BIAS);
  float*    gaml  = (float*)(smem + OFF_GAM);
  float*    betl  = (float*)(smem + OFF_BET);
  float*    b4l   = (float*)(smem + OFF_B4);

  const int tid = threadIdx.x;

  // ---- one-time weight staging (f16 for WMMA B operands; params fp32) ----
  for (int i = tid; i < 128 * 64;  i += 128) w1l[i] = (_Float16)W1[i];
  for (int i = tid; i < 128 * 128; i += 128) w2l[i] = (_Float16)W2[i];
  for (int i = tid; i < 64 * 128;  i += 128) w3l[i] = (_Float16)W3[i];
  if (tid < 64) { w0l[tid] = W0[tid]; w4l[tid] = W4[tid]; }
  for (int i = tid; i < 64;  i += 128) { biasl[i]       = b0[i]; gaml[i]       = g0[i]; betl[i]       = be0[i]; }
  for (int i = tid; i < 128; i += 128) { biasl[64 + i]  = b1[i]; gaml[64 + i]  = g1[i]; betl[64 + i]  = be1[i];
                                         biasl[192 + i] = b2[i]; gaml[192 + i] = g2[i]; betl[192 + i] = be2[i]; }
  for (int i = tid; i < 64;  i += 128) { biasl[320 + i] = b3[i]; gaml[320 + i] = g3[i]; betl[320 + i] = be3[i]; }
  if (tid == 0) b4l[0] = b4[0];
  __syncthreads();   // only block-wide sync needed; tile state is wave-private

  const int wave = tid >> 5;
  const int lane = tid & 31;
  float*    U0 = (float*)(smem + OFF_U) + wave * 3 * (16 * TW);
  float*    U1 = U0 + 16 * TW;
  float*    U2 = U1 + 16 * TW;
  _Float16* A0 = (_Float16*)(smem + OFF_A) + wave * 3 * (16 * TW);
  _Float16* A1 = A0 + 16 * TW;
  _Float16* A2 = A1 + 16 * TW;

  // Persistent over TILES sample tiles: amortizes weight staging; all control
  // flow below is wave-uniform, so EXEC stays all-ones at every WMMA.
  for (int it = 0; it < TILES; ++it) {
    const int sbase = (blockIdx.x * TILES + it) * 64 + wave * 16;

    // ---- layer 0 (1 -> 64): u0 = W0*t (+b0 in tanh_ln), u1 = W0, u2 = 0 ----
    {
      const int m = lane & 15, hh = lane >> 4;
      const float tm = t[sbase + m];
      #pragma unroll 4
      for (int j = hh * 32; j < hh * 32 + 32; ++j) {
        const float w = w0l[j];
        const int o = m * TW + j;
        U0[o] = w * tm; U1[o] = w; U2[o] = 0.0f;
      }
    }
    tanh_ln<64>(U0, U1, U2, A0, A1, A2, biasl + 0,   gaml + 0,   betl + 0,   lane);

    gemm3<64, 128>(w1l, A0, A1, A2, U0, U1, U2, lane);            // layer 1
    tanh_ln<128>(U0, U1, U2, A0, A1, A2, biasl + 64,  gaml + 64,  betl + 64,  lane);

    gemm3<128, 128>(w2l, A0, A1, A2, U0, U1, U2, lane);           // layer 2
    tanh_ln<128>(U0, U1, U2, A0, A1, A2, biasl + 192, gaml + 192, betl + 192, lane);

    gemm3<128, 64>(w3l, A0, A1, A2, U0, U1, U2, lane);            // layer 3
    tanh_ln<64>(U0, U1, U2, A0, A1, A2, biasl + 320, gaml + 320, betl + 320, lane);

    // ---- head (64 -> 1): fp32 dot per sample, pair-lane reduction ----
    {
      const int m = lane & 15, hh = lane >> 4;
      float s0 = 0.f, s1 = 0.f, s2 = 0.f;
      #pragma unroll 4
      for (int j = hh * 32; j < hh * 32 + 32; ++j) {
        const float w = w4l[j];
        const int o = m * TW + j;
        s0 += w * U0[o]; s1 += w * U1[o]; s2 += w * U2[o];
      }
      s0 += __shfl_xor(s0, 16); s1 += __shfl_xor(s1, 16); s2 += __shfl_xor(s2, 16);
      if (hh == 0) {
        const int s = sbase + m;
        out[s]         = s0 + b4l[0];   // x
        out[B + s]     = s1;            // v = dx/dt
        out[2 * B + s] = s2;            // a = d2x/dt2
      }
    }
  }
}

extern "C" void kernel_launch(void* const* d_in, const int* in_sizes, int n_in,
                              void* d_out, int out_size, void* d_ws, size_t ws_size,
                              hipStream_t stream) {
  (void)n_in; (void)out_size; (void)d_ws; (void)ws_size;
  const float* t   = (const float*)d_in[0];
  const float* W0  = (const float*)d_in[1];
  const float* b0  = (const float*)d_in[2];
  const float* g0  = (const float*)d_in[3];
  const float* be0 = (const float*)d_in[4];
  const float* W1  = (const float*)d_in[5];
  const float* b1  = (const float*)d_in[6];
  const float* g1  = (const float*)d_in[7];
  const float* be1 = (const float*)d_in[8];
  const float* W2  = (const float*)d_in[9];
  const float* b2  = (const float*)d_in[10];
  const float* g2  = (const float*)d_in[11];
  const float* be2 = (const float*)d_in[12];
  const float* W3  = (const float*)d_in[13];
  const float* b3  = (const float*)d_in[14];
  const float* g3  = (const float*)d_in[15];
  const float* be3 = (const float*)d_in[16];
  const float* W4  = (const float*)d_in[17];
  const float* b4  = (const float*)d_in[18];
  float* out = (float*)d_out;

  const int B = in_sizes[0];          // 131072; divisible by 64*TILES = 256
  const int blocks = B / (64 * TILES);

  pinn_jet2_kernel<<<blocks, 128, SMEM_BYTES, stream>>>(
      t, W0, b0, g0, be0, W1, b1, g1, be1, W2, b2, g2, be2,
      W3, b3, g3, be3, W4, b4, out, B);
}
